// RegionProposalNetwork_45148696215860
// MI455X (gfx1250) — matrix operand order repaired and checked by reading, not compiled, not run on hardware
//
#include <hip/hip_runtime.h>
#include <math.h>

// ---------------------------------------------------------------------------
// RPN (Faster R-CNN) for MI455X / gfx1250.
// Hot op: 3x3 conv (1,1024,96,96)->(1,512,96,96) = 87 GFLOP, compute bound.
// Implicit GEMM over 9 shift-phases with V_WMMA_F32_16X16X32_BF16 using a
// bf16 hi/lo split of the fp32 operands (3 WMMAs per K-chunk) for ~fp32
// accuracy at bf16-matrix throughput. Workgroup tile = 128 co x 96 cols
// (full row) so each wave issues 18 WMMAs per 32-ci chunk against one shared
// LDS gather. BN+ReLU fused into the WMMA epilogue.
// ---------------------------------------------------------------------------

typedef __attribute__((ext_vector_type(16))) __bf16 bf16x16;
typedef __attribute__((ext_vector_type(8)))  float  f32x8;

#define NSP   9216      // 96*96 spatial
#define CI    1024
#define CO    512
#define NANCH 138240    // 9216*15
#define PRE_NMS 6000
#define POST_NMS 300

// ---- workspace layout (bytes) ----
#define WS_WPH    0u                    // bf16 [9][512][1024]  9,437,184 B
#define WS_WPL    9437184u              // bf16 same
#define WS_XMID   18874368u             // f32  [512][9216]    18,874,368 B
#define WS_OUT90  37748736u             // f32  [90][9216]      3,317,760 B
#define WS_SCORES 41066496u             // f32  [138240]          552,960 B
#define WS_BOXES  41619456u             // float4 [138240]      2,211,840 B
#define WS_WC     43831296u             // f32  [90][512]         184,320 B
#define WS_BC     44016128u             // f32  [90]
#define WS_S      44016640u             // f32  [512]
#define WS_T      44018688u             // f32  [512]
#define WS_HIST   44020736u             // i32  [1024]
#define WS_FHIST  44024832u             // i32  [1024]
#define WS_CTL    44028928u

struct Ctl { int coarse_b; int need; int skip; float lo; float thresh; };

// ---------------------------------------------------------------------------
// K1: prepack conv1 weights into [phase][co][ci] bf16 hi + lo
// ---------------------------------------------------------------------------
__global__ __launch_bounds__(256) void k_prep_w(const float* __restrict__ w,
                                                __bf16* __restrict__ wph,
                                                __bf16* __restrict__ wpl) {
  int o = blockIdx.x * 256 + threadIdx.x;            // o < 9*512*1024
  int r  = o >> 19;
  int co = (o >> 10) & 511;
  int ci = o & 1023;
  float v = w[co * 9216 + ci * 9 + r];
  __bf16 h = (__bf16)v;
  wph[o] = h;
  wpl[o] = (__bf16)(v - (float)h);
}

// ---------------------------------------------------------------------------
// K2: fold conv bias + BN into per-channel affine; build combined 1x1 weights;
//     zero histograms / control block (must run every call: graph replay).
// ---------------------------------------------------------------------------
__global__ __launch_bounds__(1024) void k_setup(const float* __restrict__ conv1_b,
                                                const float* __restrict__ gam,
                                                const float* __restrict__ bet,
                                                const float* __restrict__ mean,
                                                const float* __restrict__ var,
                                                const float* __restrict__ score_w,
                                                const float* __restrict__ score_b,
                                                const float* __restrict__ bbox_w,
                                                const float* __restrict__ bbox_b,
                                                float* __restrict__ S, float* __restrict__ T,
                                                float* __restrict__ Wc, float* __restrict__ Bc,
                                                int* __restrict__ hist, int* __restrict__ fhist,
                                                Ctl* ctl) {
  int t = threadIdx.x;
  if (t < 512) {
    float s = gam[t] / sqrtf(var[t] + 1e-5f);
    S[t] = s;
    T[t] = (conv1_b[t] - mean[t]) * s + bet[t];
  }
  for (int i = t; i < 90 * 512; i += 1024) {
    int oc = i >> 9;
    Wc[i] = (oc < 30) ? score_w[i] : bbox_w[i - 30 * 512];
  }
  if (t < 90) Bc[t] = (t < 30) ? score_b[t] : bbox_b[t - 30];
  if (t < 1024) { hist[t] = 0; fhist[t] = 0; }
  if (t == 0) { ctl->coarse_b = -1; ctl->need = 0; ctl->skip = 1; ctl->lo = 0.f; ctl->thresh = 0.f; }
}

// ---------------------------------------------------------------------------
// K3: implicit-GEMM 3x3 conv + BN + ReLU via bf16 hi/lo split WMMA.
// Grid: (96 image rows, 4 co-blocks of 128). Block 256 = 8 waves; wave wv
// owns m-tile (16 co) x all six 16-col n-tiles -> 6 accumulators, 18 WMMAs
// per 32-ci chunk.
// ---------------------------------------------------------------------------
__device__ inline bf16x16 ldfrag2(const __bf16* p0, const __bf16* p1) {
  union { bf16x16 v; uint4 q[2]; } u;
  u.q[0] = *reinterpret_cast<const uint4*>(p0);
  u.q[1] = *reinterpret_cast<const uint4*>(p1);
  return u.v;
}

__global__ __launch_bounds__(256) void k_conv3x3(const float* __restrict__ feat,
                                                 const __bf16* __restrict__ wph,
                                                 const __bf16* __restrict__ wpl,
                                                 const float* __restrict__ S,
                                                 const float* __restrict__ T,
                                                 float* __restrict__ xmid) {
  const int ytile = blockIdx.x;              // 0..95, full 96-wide row
  const int co0 = blockIdx.y * 128;
  const int tid = threadIdx.x;
  const int wv = tid >> 5, lane = tid & 31;
  const int half = lane >> 4, l16 = lane & 15;

  __shared__ __align__(16) __bf16 Bh[96 * 40 + 8];
  __shared__ __align__(16) __bf16 Bl[96 * 40 + 8];

  f32x8 acc[6];
#pragma unroll
  for (int nt = 0; nt < 6; ++nt) acc[nt] = (f32x8){};

  const int xl   = tid & 31;   // fill: lane-x within a 32-col pass
  const int ci_l = (tid >> 5) * 4;   // fill: 4 consecutive ci per thread

  for (int ph = 0; ph < 9; ++ph) {
    const int dy = ph / 3 - 1, dx = ph % 3 - 1;
    const int yy = ytile + dy;
    if (yy < 0 || yy >= 96) continue;
    const float* frow = feat + yy * 96;
    for (int ci0 = 0; ci0 < CI; ci0 += 32) {
      __syncthreads();
      // --- gather + fp32->bf16 hi/lo split into LDS, B layout [n][k] ---
      // Thread handles (x, ci..ci+3): loads coalesced across lanes in x,
      // LDS stores are contiguous b64. Edge cols: clamp address + select.
      {
        const float* fp0 = frow + (ci0 + ci_l) * NSP;
#pragma unroll
        for (int p = 0; p < 3; ++p) {
          int x  = p * 32 + xl;
          int xx = x + dx;
          int inb = (xx >= 0) & (xx < 96);
          int xc = min(max(xx, 0), 95);
          union { __bf16 e[4]; uint2 q; } uh, ul;
#pragma unroll
          for (int j = 0; j < 4; ++j) {
            float t = fp0[j * NSP + xc];
            t = inb ? t : 0.0f;
            __bf16 h = (__bf16)t;
            uh.e[j] = h;
            ul.e[j] = (__bf16)(t - (float)h);
          }
          *reinterpret_cast<uint2*>(&Bh[x * 40 + ci_l]) = uh.q;
          *reinterpret_cast<uint2*>(&Bl[x * 40 + ci_l]) = ul.q;
        }
      }
      __syncthreads();
      // A fragments direct from global (L2-resident weights).
      // 16-bit A 16x32 layout: lane<16 -> K {0..7,16..23}, lane>=16 -> +8.
      const int m = co0 + wv * 16 + l16;
      const long ab = (long)(ph * 512 + m) * 1024 + ci0 + half * 8;
      bf16x16 ahf = ldfrag2(wph + ab, wph + ab + 16);
      bf16x16 alf = ldfrag2(wpl + ab, wpl + ab + 16);
      // B fragments from LDS. 16-bit B 32x16 layout: lane L -> N=L%16,
      // K {0..7,16..23} for L<16, +8 for L>=16 (matches A).
#pragma unroll
      for (int nt = 0; nt < 6; ++nt) {
        const int nrow = (nt * 16 + l16) * 40 + half * 8;
        bf16x16 bhf = ldfrag2(&Bh[nrow], &Bh[nrow + 16]);
        bf16x16 blf = ldfrag2(&Bl[nrow], &Bl[nrow + 16]);
        // hi*hi + hi*lo + lo*hi  (lo*lo negligible)
        acc[nt] = __builtin_amdgcn_wmma_f32_16x16x32_bf16(false, ahf, false, bhf, (short)0, acc[nt], false, false);
        acc[nt] = __builtin_amdgcn_wmma_f32_16x16x32_bf16(false, ahf, false, blf, (short)0, acc[nt], false, false);
        acc[nt] = __builtin_amdgcn_wmma_f32_16x16x32_bf16(false, alf, false, bhf, (short)0, acc[nt], false, false);
      }
    }
  }
  // epilogue: fused BN + ReLU. D layout: lane<16 M=r, lane>=16 M=8+r; N=l16.
  const int co = co0 + wv * 16 + half * 8;
#pragma unroll
  for (int nt = 0; nt < 6; ++nt) {
    const int nout = ytile * 96 + nt * 16 + l16;
#pragma unroll
    for (int r = 0; r < 8; ++r) {
      float v = fmaf(acc[nt][r], S[co + r], T[co + r]);
      xmid[(long)(co + r) * NSP + nout] = fmaxf(v, 0.0f);
    }
  }
}

// ---------------------------------------------------------------------------
// K4: 1x1 convs (90 channels = 30 score + 60 bbox) as tiled GEMM, K=512.
// ---------------------------------------------------------------------------
__global__ __launch_bounds__(256) void k_conv1x1(const float* __restrict__ xmid,
                                                 const float* __restrict__ Wc,
                                                 const float* __restrict__ Bc,
                                                 float* __restrict__ out90) {
  const int n0 = blockIdx.x * 64;
  const int tid = threadIdx.x;
  const int nl = tid & 63, g = tid >> 6;       // g in [0,4)
  __shared__ float sX[32][64];
  float acc[23];
#pragma unroll
  for (int j = 0; j < 23; ++j) acc[j] = 0.0f;

  for (int ci0 = 0; ci0 < 512; ci0 += 32) {
    __syncthreads();
#pragma unroll
    for (int k = 0; k < 8; ++k) {
      int ci = g + 4 * k;
      sX[ci][nl] = xmid[(long)(ci0 + ci) * NSP + n0 + nl];
    }
    __syncthreads();
    for (int ci = 0; ci < 32; ++ci) {
      float xv = sX[ci][nl];
#pragma unroll
      for (int j = 0; j < 23; ++j) {
        int oc = g + 4 * j;
        if (oc < 90) acc[j] = fmaf(Wc[oc * 512 + ci0 + ci], xv, acc[j]);
      }
    }
  }
#pragma unroll
  for (int j = 0; j < 23; ++j) {
    int oc = g + 4 * j;
    if (oc < 90) out90[(long)oc * NSP + n0 + nl] = acc[j] + Bc[oc];
  }
}

// ---------------------------------------------------------------------------
// K5: per-anchor score (pairwise softmax == sigmoid), bbox transform, clip,
//     min-size filter; coarse 1024-bin histogram of scores.
// ---------------------------------------------------------------------------
__global__ __launch_bounds__(256) void k_prop(const float* __restrict__ out90,
                                              const float* __restrict__ im_info,
                                              float* __restrict__ scores,
                                              float4* __restrict__ boxes,
                                              int* __restrict__ hist) {
  int idx = blockIdx.x * 256 + threadIdx.x;   // < 138240
  int a = idx % 15, n = idx / 15;
  int y = n / 96, x = n % 96;
  int ri = a / 5, si = a % 5;
  float ratio = (ri == 0) ? 0.5f : (ri == 1 ? 1.0f : 2.0f);
  float scale = (float)(2 << si);
  // _generate_anchors (ratio enum, then scale enum), banker's rint matches np
  float ws0 = rintf(sqrtf(256.0f / ratio));
  float hs0 = rintf(ws0 * ratio);
  float aw = ws0 * scale, ah = hs0 * scale;   // scaled w/h; centers stay 7.5
  float acx = 7.5f + x * 16.0f;
  float acy = 7.5f + y * 16.0f;

  float d0 = out90[(long)(30 + 4 * a + 0) * NSP + n];
  float d1 = out90[(long)(30 + 4 * a + 1) * NSP + n];
  float d2 = out90[(long)(30 + 4 * a + 2) * NSP + n];
  float d3 = out90[(long)(30 + 4 * a + 3) * NSP + n];
  float px = d0 * aw + acx, py = d1 * ah + acy;
  float pw = expf(d2) * aw, phh = expf(d3) * ah;

  float imH = im_info[0], imW = im_info[1], imS = im_info[2];
  float x1 = fminf(fmaxf(px - 0.5f * pw, 0.f), imW - 1.f);
  float y1 = fminf(fmaxf(py - 0.5f * phh, 0.f), imH - 1.f);
  float x2 = fminf(fmaxf(px + 0.5f * pw - 1.f, 0.f), imW - 1.f);
  float y2 = fminf(fmaxf(py + 0.5f * phh - 1.f, 0.f), imH - 1.f);

  float c0 = out90[(long)a * NSP + n];
  float c1 = out90[(long)(15 + a) * NSP + n];
  float sc = 1.0f / (1.0f + expf(c0 - c1));
  float ms = 16.0f * imS;
  if (!(((x2 - x1 + 1.f) >= ms) && ((y2 - y1 + 1.f) >= ms))) sc = -1.0f;

  scores[idx] = sc;
  boxes[idx] = make_float4(x1, y1, x2, y2);
  if (sc >= 0.f) {
    int b = min(1023, (int)(sc * 1024.0f));
    atomicAdd(&hist[b], 1);
  }
}

// K6: coarse scan from the top -> boundary bin + remaining count
__global__ void k_coarse(const int* __restrict__ hist, Ctl* ctl) {
  int cum = 0, found = 0;
  for (int b = 1023; b >= 0; --b) {
    int c = hist[b];
    if (cum + c >= PRE_NMS) {
      ctl->coarse_b = b; ctl->need = PRE_NMS - cum;
      ctl->lo = b * (1.0f / 1024.0f);
      ctl->thresh = ctl->lo; ctl->skip = 0; found = 1; break;
    }
    cum += c;
  }
  if (!found) { ctl->skip = 1; ctl->thresh = 0.0f; ctl->lo = 0.0f; ctl->coarse_b = -1; ctl->need = 0; }
}

// K7: fine 1024-bin histogram inside boundary bin (resolution 2^-20)
__global__ __launch_bounds__(256) void k_fine(const float* __restrict__ scores,
                                              const Ctl* __restrict__ ctl,
                                              int* __restrict__ fhist) {
  if (ctl->skip) return;
  int idx = blockIdx.x * 256 + threadIdx.x;
  float s = scores[idx];
  if (s >= 0.f) {
    int b = min(1023, (int)(s * 1024.0f));
    if (b == ctl->coarse_b) {
      int fb = (int)((s - ctl->lo) * 1048576.0f);
      fb = max(0, min(1023, fb));
      atomicAdd(&fhist[fb], 1);
    }
  }
}

// K8: fine scan -> final threshold (count(score>=thresh) ~= 6000)
__global__ void k_fsel(const int* __restrict__ fhist, Ctl* ctl) {
  if (ctl->skip) return;
  int cum = 0; float th = ctl->lo;
  for (int fb = 1023; fb >= 0; --fb) {
    int c = fhist[fb];
    if (cum + c >= ctl->need) { th = ctl->lo + fb * (1.0f / 1048576.0f); break; }
    cum += c;
  }
  ctl->thresh = th;
}

// ---------------------------------------------------------------------------
// K9: greedy NMS, 300 sequential argmax+suppress iterations (single WG so
// everything stays ordered/deterministic; candidate set = score >= thresh).
// ---------------------------------------------------------------------------
__global__ __launch_bounds__(1024) void k_nms(float* __restrict__ scores,
                                              const float4* __restrict__ boxes,
                                              const Ctl* __restrict__ ctl,
                                              float* __restrict__ rois) {
  __shared__ float rv[1024];
  __shared__ int   rj[1024];
  __shared__ float4 spick;
  __shared__ int sok;
  const int tid = threadIdx.x;
  const float th = ctl->thresh;

  for (int it = 0; it < POST_NMS; ++it) {
    float best = -1e30f; int bi = 0x7fffffff;
    for (int i = tid; i < NANCH; i += 1024) {
      float s = scores[i];
      if (s >= th && s > best) { best = s; bi = i; }
    }
    rv[tid] = best; rj[tid] = bi;
    __syncthreads();
    for (int st = 512; st > 0; st >>= 1) {
      if (tid < st) {
        if (rv[tid + st] > rv[tid] ||
            (rv[tid + st] == rv[tid] && rj[tid + st] < rj[tid])) {
          rv[tid] = rv[tid + st]; rj[tid] = rj[tid + st];
        }
      }
      __syncthreads();
    }
    if (tid == 0) {
      int ok = (rv[0] >= th) && (rj[0] < NANCH);
      sok = ok;
      float4 pb = ok ? boxes[rj[0]] : make_float4(0.f, 0.f, 0.f, 0.f);
      spick = pb;
      if (ok) scores[rj[0]] = -2.0f;
      rois[it * 5 + 0] = 0.f;
      rois[it * 5 + 1] = pb.x; rois[it * 5 + 2] = pb.y;
      rois[it * 5 + 3] = pb.z; rois[it * 5 + 4] = pb.w;
    }
    __syncthreads();
    if (sok) {
      float4 pb = spick;
      float areaj = (pb.z - pb.x + 1.f) * (pb.w - pb.y + 1.f);
      for (int i = tid; i < NANCH; i += 1024) {
        float s = scores[i];
        if (s >= th) {
          float4 b = boxes[i];
          float xx1 = fmaxf(pb.x, b.x), yy1 = fmaxf(pb.y, b.y);
          float xx2 = fminf(pb.z, b.z), yy2 = fminf(pb.w, b.w);
          float iw = fmaxf(0.f, xx2 - xx1 + 1.f);
          float ih = fmaxf(0.f, yy2 - yy1 + 1.f);
          float inter = iw * ih;
          float areai = (b.z - b.x + 1.f) * (b.w - b.y + 1.f);
          float iou = inter / (areaj + areai - inter);
          if (iou > 0.7f) scores[i] = -2.0f;
        }
      }
    }
    __syncthreads();
  }
}

// ---------------------------------------------------------------------------
extern "C" void kernel_launch(void* const* d_in, const int* in_sizes, int n_in,
                              void* d_out, int out_size, void* d_ws, size_t ws_size,
                              hipStream_t stream) {
  const float* feat     = (const float*)d_in[0];
  const float* im_info  = (const float*)d_in[1];
  const float* conv1_w  = (const float*)d_in[2];
  const float* conv1_b  = (const float*)d_in[3];
  const float* bn_gamma = (const float*)d_in[4];
  const float* bn_beta  = (const float*)d_in[5];
  const float* bn_mean  = (const float*)d_in[6];
  const float* bn_var   = (const float*)d_in[7];
  const float* score_w  = (const float*)d_in[8];
  const float* score_b  = (const float*)d_in[9];
  const float* bbox_w   = (const float*)d_in[10];
  const float* bbox_b   = (const float*)d_in[11];

  char* ws = (char*)d_ws;
  __bf16* wph   = (__bf16*)(ws + WS_WPH);
  __bf16* wpl   = (__bf16*)(ws + WS_WPL);
  float*  xmid  = (float*)(ws + WS_XMID);
  float*  out90 = (float*)(ws + WS_OUT90);
  float*  sco   = (float*)(ws + WS_SCORES);
  float4* box   = (float4*)(ws + WS_BOXES);
  float*  Wc    = (float*)(ws + WS_WC);
  float*  Bc    = (float*)(ws + WS_BC);
  float*  S     = (float*)(ws + WS_S);
  float*  T     = (float*)(ws + WS_T);
  int*    hist  = (int*)(ws + WS_HIST);
  int*    fhist = (int*)(ws + WS_FHIST);
  Ctl*    ctl   = (Ctl*)(ws + WS_CTL);

  float* rois = ((float*)d_out) + (size_t)CI * NSP;   // 9,437,184 floats in

  // pass-through output #1: features
  hipMemcpyAsync(d_out, feat, (size_t)CI * NSP * sizeof(float),
                 hipMemcpyDeviceToDevice, stream);

  k_prep_w<<<dim3(9 * 512 * 1024 / 256), dim3(256), 0, stream>>>(conv1_w, wph, wpl);
  k_setup<<<dim3(1), dim3(1024), 0, stream>>>(conv1_b, bn_gamma, bn_beta, bn_mean,
                                              bn_var, score_w, score_b, bbox_w,
                                              bbox_b, S, T, Wc, Bc, hist, fhist, ctl);
  k_conv3x3<<<dim3(96, 4), dim3(256), 0, stream>>>(feat, wph, wpl, S, T, xmid);
  k_conv1x1<<<dim3(NSP / 64), dim3(256), 0, stream>>>(xmid, Wc, Bc, out90);
  k_prop<<<dim3(NANCH / 256), dim3(256), 0, stream>>>(out90, im_info, sco, box, hist);
  k_coarse<<<dim3(1), dim3(1), 0, stream>>>(hist, ctl);
  k_fine<<<dim3(NANCH / 256), dim3(256), 0, stream>>>(sco, ctl, fhist);
  k_fsel<<<dim3(1), dim3(1), 0, stream>>>(fhist, ctl);
  k_nms<<<dim3(1), dim3(1024), 0, stream>>>(sco, box, ctl, rois);
}
